// VQcoorExtractor_86827058856706
// MI455X (gfx1250) — compile-verified, matrix operand order checked by reading, not run on hardware
//
#include <hip/hip_runtime.h>

// ---------------------------------------------------------------------------
// Problem constants (from reference)
// ---------------------------------------------------------------------------
#define BB   16
#define NN   512
#define DD   256
#define DA   128
#define KK   64
#define HH   1024
#define MM   (BB * NN)            // 8192 flattened rows
#define PSI_ELEMS ((long long)MM * KK * 2 * DD)   // 268,435,456

// ---------------------------------------------------------------------------
// Vector types
// ---------------------------------------------------------------------------
typedef __attribute__((ext_vector_type(4)))  float          f4;
typedef __attribute__((ext_vector_type(8)))  float          vf8;
typedef __attribute__((ext_vector_type(8)))  float          v8f;
typedef __attribute__((ext_vector_type(8)))  __bf16         v8bf;
typedef __attribute__((ext_vector_type(16))) __bf16         v16bf;
typedef __attribute__((ext_vector_type(8)))  unsigned short us8;
typedef __attribute__((ext_vector_type(16))) unsigned short us16;
typedef int v4i __attribute__((vector_size(16)));           // builtin's pointee type

#define AS1 __attribute__((address_space(1)))
#define AS3 __attribute__((address_space(3)))

// CDNA5 async global->LDS staging (ASYNCcnt) if the toolchain exposes it.
#if __has_builtin(__builtin_amdgcn_global_load_async_to_lds_b128) && \
    __has_builtin(__builtin_amdgcn_s_wait_asynccnt)
#define HAVE_ASYNC 1
#else
#define HAVE_ASYNC 0
#endif

// Load one 16x32 bf16 WMMA fragment (ISA layout: lanes 0-15 hold K{0..7,16..23},
// lanes 16-31 hold K{8..15,24..31}) from an LDS row of bf16 (caller passes
// &row[halfk]); two 16B LDS reads.
__device__ __forceinline__ v16bf ld_frag_bf16(const __bf16* p) {
  const us8* q = (const us8*)p;
  union { us16 u; v16bf v; } r;
  r.u = __builtin_shufflevector(q[0], q[2], 0,1,2,3,4,5,6,7,8,9,10,11,12,13,14,15);
  return r.v;
}

#if HAVE_ASYNC
// Same fragment but from an f32 LDS row: 2x32B LDS reads + packed cvt to bf16.
__device__ __forceinline__ v16bf ld_frag_f32(const float* p) {
  v8bf lo = __builtin_convertvector(*(const vf8*)p, v8bf);
  v8bf hi = __builtin_convertvector(*(const vf8*)(p + 16), v8bf);
  return __builtin_shufflevector(lo, hi, 0,1,2,3,4,5,6,7,8,9,10,11,12,13,14,15);
}
#endif

// ---------------------------------------------------------------------------
// bf16 WMMA GEMM:  C[M,N] = act( A[M,K] @ B[K,N] + bias )
//   - 64x64 block tile, 128 threads (4 waves), wave = 32x32 (2x2 frags)
//   - double-buffered LDS, software-pipelined staging
//   - A tile staged via GLOBAL_LOAD_ASYNC_TO_LDS_B128 when available
//   - TRANSB: B given as [N,K] row-major (use B^T)
//   - batched via blockIdx.z with element strides sA/sB/sC
//   - ACT: 0 = identity, 1 = leaky_relu(0.01)
// ---------------------------------------------------------------------------
template <int ACT, bool TRANSB>
__global__ __launch_bounds__(128) void gemm_bf16_wmma(
    const float* __restrict__ A,  int lda, long long sA,
    const float* __restrict__ Bm, int ldb, long long sB,
    const float* __restrict__ bias,
    float* __restrict__ C,        int ldc, long long sC,
    int Kc)
{
#if HAVE_ASYNC
  __shared__ __align__(16) float  As32[2][64 * 36];   // f32 tiles, stride 36
#else
  __shared__ __align__(16) __bf16 As[2][64 * 40];     // bf16 tiles, stride 40
#endif
  __shared__ __align__(16) __bf16 Bs[2][64 * 40];     // Bs[n][k], bf16

  const int t    = threadIdx.x;
  const int lane = t & 31;
  const int wave = t >> 5;
  const int wm   = (wave >> 1) * 32;
  const int wn   = (wave & 1) * 32;
  const int rowBase = blockIdx.y * 64;
  const int colBase = blockIdx.x * 64;

  A  += (long long)blockIdx.z * sA;
  Bm += (long long)blockIdx.z * sB;
  C  += (long long)blockIdx.z * sC;

  v8f acc00 = {}, acc01 = {}, acc10 = {}, acc11 = {};

  const int r15   = lane & 15;
  const int halfk = (lane >> 4) * 8;

  f4 ra[4], rb[4];

  // ---- staging helpers ---------------------------------------------------
  auto issueA = [&](int k0, int bsel) {
    #pragma unroll
    for (int i = 0; i < 4; ++i) {
      int e = t + i * 128, row = e >> 3, c4 = (e & 7) * 4;
      const float* g = &A[(long long)(rowBase + row) * lda + k0 + c4];
#if HAVE_ASYNC
      __builtin_amdgcn_global_load_async_to_lds_b128(
          (AS1 v4i*)g, (AS3 v4i*)&As32[bsel][row * 36 + c4], 0, 0);
#else
      (void)bsel;
      ra[i] = *(const f4*)g;
#endif
    }
  };
  auto issueB = [&](int k0) {
    #pragma unroll
    for (int i = 0; i < 4; ++i) {
      int e = t + i * 128;
      if constexpr (TRANSB) {
        int n = e >> 3, c4 = (e & 7) * 4;
        rb[i] = *(const f4*)&Bm[(long long)(colBase + n) * ldb + k0 + c4];
      } else {
        int kk = e >> 4, c4 = (e & 15) * 4;
        rb[i] = *(const f4*)&Bm[(long long)(k0 + kk) * ldb + colBase + c4];
      }
    }
  };
  auto commit = [&](int bsel) {
    #pragma unroll
    for (int i = 0; i < 4; ++i) {
      int e = t + i * 128;
      if constexpr (TRANSB) {
        int n = e >> 3, c4 = (e & 7) * 4;
        __bf16* d = &Bs[bsel][n * 40 + c4];
        d[0] = (__bf16)rb[i][0]; d[1] = (__bf16)rb[i][1];
        d[2] = (__bf16)rb[i][2]; d[3] = (__bf16)rb[i][3];
      } else {
        int kk = e >> 4, c4 = (e & 15) * 4;
        Bs[bsel][(c4 + 0) * 40 + kk] = (__bf16)rb[i][0];
        Bs[bsel][(c4 + 1) * 40 + kk] = (__bf16)rb[i][1];
        Bs[bsel][(c4 + 2) * 40 + kk] = (__bf16)rb[i][2];
        Bs[bsel][(c4 + 3) * 40 + kk] = (__bf16)rb[i][3];
      }
    }
#if HAVE_ASYNC
    __builtin_amdgcn_s_wait_asynccnt(0);
#else
    #pragma unroll
    for (int i = 0; i < 4; ++i) {
      int e = t + i * 128, row = e >> 3, c4 = (e & 7) * 4;
      __bf16* d = &As[bsel][row * 40 + c4];
      d[0] = (__bf16)ra[i][0]; d[1] = (__bf16)ra[i][1];
      d[2] = (__bf16)ra[i][2]; d[3] = (__bf16)ra[i][3];
    }
#endif
  };

  // ---- prologue ----
  issueA(0, 0);
  issueB(0);
  commit(0);
  __syncthreads();

  int buf = 0;
  for (int k0 = 0; k0 < Kc; k0 += 32) {
    const bool more = (k0 + 32) < Kc;
    if (more) { issueA(k0 + 32, buf ^ 1); issueB(k0 + 32); }

    // ---- fragments + WMMA on current buffer ----
    v16bf a0, a1, b0, b1;
#if HAVE_ASYNC
    a0 = ld_frag_f32(&As32[buf][(wm + r15) * 36 + halfk]);
    a1 = ld_frag_f32(&As32[buf][(wm + 16 + r15) * 36 + halfk]);
#else
    a0 = ld_frag_bf16(&As[buf][(wm + r15) * 40 + halfk]);
    a1 = ld_frag_bf16(&As[buf][(wm + 16 + r15) * 40 + halfk]);
#endif
    b0 = ld_frag_bf16(&Bs[buf][(wn + r15) * 40 + halfk]);
    b1 = ld_frag_bf16(&Bs[buf][(wn + 16 + r15) * 40 + halfk]);

    acc00 = __builtin_amdgcn_wmma_f32_16x16x32_bf16(false, a0, false, b0, (short)0, acc00, false, false);
    acc01 = __builtin_amdgcn_wmma_f32_16x16x32_bf16(false, a0, false, b1, (short)0, acc01, false, false);
    acc10 = __builtin_amdgcn_wmma_f32_16x16x32_bf16(false, a1, false, b0, (short)0, acc10, false, false);
    acc11 = __builtin_amdgcn_wmma_f32_16x16x32_bf16(false, a1, false, b1, (short)0, acc11, false, false);

    if (more) commit(buf ^ 1);
    __syncthreads();
    buf ^= 1;
  }

  // ---- epilogue: C/D layout -> lanes 0-15: M=v, lanes 16-31: M=v+8 ----
  const int hi = (lane >> 4) * 8;
  #pragma unroll
  for (int ij = 0; ij < 4; ++ij) {
    int ti = ij >> 1, tj = ij & 1;
    v8f a = (ij == 0) ? acc00 : (ij == 1) ? acc01 : (ij == 2) ? acc10 : acc11;
    int col = colBase + wn + tj * 16 + r15;
    int rb_ = rowBase + wm + ti * 16 + hi;
    float bv = bias ? bias[col] : 0.f;
    #pragma unroll
    for (int v = 0; v < 8; ++v) {
      float o = a[v] + bv;
      if (ACT == 1) o = (o > 0.f) ? o : 0.01f * o;
      C[(long long)(rb_ + v) * ldc + col] = o;
    }
  }
}

// ---------------------------------------------------------------------------
// Softmax over rows of S [8192 x 512], with scale applied first
// ---------------------------------------------------------------------------
__global__ __launch_bounds__(256) void softmax_rows(float* __restrict__ S, float scale) {
  __shared__ float red[256];
  const int t = threadIdx.x;
  float* row = S + (long long)blockIdx.x * NN;
  float s0 = row[t] * scale, s1 = row[t + 256] * scale;
  red[t] = fmaxf(s0, s1);
  __syncthreads();
  for (int st = 128; st > 0; st >>= 1) { if (t < st) red[t] = fmaxf(red[t], red[t + st]); __syncthreads(); }
  float mx = red[0];
  __syncthreads();
  float e0 = expf(s0 - mx), e1 = expf(s1 - mx);
  red[t] = e0 + e1;
  __syncthreads();
  for (int st = 128; st > 0; st >>= 1) { if (t < st) red[t] += red[t + st]; __syncthreads(); }
  float inv = 1.f / red[0];
  row[t] = e0 * inv;
  row[t + 256] = e1 * inv;
}

// ---------------------------------------------------------------------------
// Fused residual + LayerNorm over D=256: out = LN(x + res) * g + b
// ---------------------------------------------------------------------------
__global__ __launch_bounds__(256) void ln_residual(
    const float* __restrict__ x, const float* __restrict__ res,
    const float* __restrict__ g, const float* __restrict__ b,
    float* __restrict__ out)
{
  __shared__ float red[256];
  const int t = threadIdx.x;
  const long long base = (long long)blockIdx.x * DD;
  float v = x[base + t] + res[base + t];
  red[t] = v;
  __syncthreads();
  for (int st = 128; st > 0; st >>= 1) { if (t < st) red[t] += red[t + st]; __syncthreads(); }
  float m = red[0] * (1.f / DD);
  __syncthreads();
  float d = v - m;
  red[t] = d * d;
  __syncthreads();
  for (int st = 128; st > 0; st >>= 1) { if (t < st) red[t] += red[t + st]; __syncthreads(); }
  float var = red[0] * (1.f / DD);
  out[base + t] = d * rsqrtf(var + 1e-5f) * g[t] + b[t];
}

// ---------------------------------------------------------------------------
// MSE accumulation: atomically adds  w * sum((a-b)^2)  into *loss
// ---------------------------------------------------------------------------
__global__ __launch_bounds__(256) void mse_acc(
    const float* __restrict__ a, const float* __restrict__ b,
    int n, float w, float* __restrict__ loss)
{
  __shared__ float red[256];
  const int t = threadIdx.x;
  float s = 0.f;
  for (int i = blockIdx.x * 256 + t; i < n; i += gridDim.x * 256) {
    float d = a[i] - b[i];
    s += d * d;
  }
  red[t] = s;
  __syncthreads();
  for (int st = 128; st > 0; st >>= 1) { if (t < st) red[t] += red[t + st]; __syncthreads(); }
  if (t == 0) atomicAdd(loss, red[0] * w);
}

// ---------------------------------------------------------------------------
// VQ: per row of lat [8192 x 256], argmin over K=64 codes, gather quant row.
// (first-min tie-break like jnp.argmin via sequential scan)
// ---------------------------------------------------------------------------
__global__ __launch_bounds__(64) void vq_kernel(
    const float* __restrict__ lat, const float* __restrict__ emb,
    float* __restrict__ quant)
{
  __shared__ float lrow[DD];
  __shared__ float dists[KK];
  __shared__ int   bidx;
  const int t = threadIdx.x;
  const long long base = (long long)blockIdx.x * DD;
  for (int i = t; i < DD; i += 64) lrow[i] = lat[base + i];
  __syncthreads();
  const float* e = emb + t * DD;
  float d = 0.f;
  #pragma unroll 4
  for (int j = 0; j < DD; j += 4) {
    f4 ev = *(const f4*)&e[j];
    f4 lv = *(const f4*)&lrow[j];
    float d0 = lv[0] - ev[0], d1 = lv[1] - ev[1];
    float d2 = lv[2] - ev[2], d3 = lv[3] - ev[3];
    d += d0 * d0 + d1 * d1 + d2 * d2 + d3 * d3;
  }
  dists[t] = d;
  __syncthreads();
  if (t == 0) {
    float best = dists[0]; int bi = 0;
    for (int kidx = 1; kidx < KK; ++kidx)
      if (dists[kidx] < best) { best = dists[kidx]; bi = kidx; }
    bidx = bi;
  }
  __syncthreads();
  const float* eb = emb + bidx * DD;
  for (int i = t; i < DD; i += 64) quant[base + i] = eb[i];
}

// ---------------------------------------------------------------------------
// psi_hid writer: out[(bn*K + k)*512 + j] = j<256 ? z_coor[bn][j] : emb[k][j-256]
// 1.07 GB streaming output -> non-temporal f4 stores; 4x unroll for MLP.
// ---------------------------------------------------------------------------
__global__ __launch_bounds__(128) void psi_kernel(
    const float* __restrict__ zc, const float* __restrict__ emb,
    float* __restrict__ out)
{
  __shared__ __align__(16) float zrow[DD];
  const int t = threadIdx.x;
  const long long row = blockIdx.x;
  for (int i = t; i < DD; i += 128) zrow[i] = zc[row * DD + i];
  __syncthreads();
  f4 vz;
  if (t < 64) vz = *(const f4*)&zrow[t * 4];
  f4* o = (f4*)(out + row * (KK * 2 * DD));
  #pragma unroll 4
  for (int k = 0; k < KK; ++k) {
    f4 vv = (t < 64) ? vz : *(const f4*)&emb[k * DD + (t - 64) * 4];
    __builtin_nontemporal_store(vv, &o[(long long)k * 128 + t]);
  }
}

// ---------------------------------------------------------------------------
// Host orchestration
// ---------------------------------------------------------------------------
extern "C" void kernel_launch(void* const* d_in, const int* in_sizes, int n_in,
                              void* d_out, int out_size, void* d_ws, size_t ws_size,
                              hipStream_t stream) {
  const float* x      = (const float*)d_in[0];
  const float* Wq_w   = (const float*)d_in[1];
  const float* Wq_b   = (const float*)d_in[2];
  const float* Wk_w   = (const float*)d_in[3];
  const float* Wk_b   = (const float*)d_in[4];
  const float* Wv_w   = (const float*)d_in[5];
  const float* Wv_b   = (const float*)d_in[6];
  const float* ln1_g  = (const float*)d_in[7];
  const float* ln1_b  = (const float*)d_in[8];
  const float* ln2_g  = (const float*)d_in[9];
  const float* ln2_b  = (const float*)d_in[10];
  const float* ffn_w1 = (const float*)d_in[11];
  const float* ffn_b1 = (const float*)d_in[12];
  const float* ffn_w2 = (const float*)d_in[13];
  const float* ffn_b2 = (const float*)d_in[14];
  const float* emb    = (const float*)d_in[15];
  const float* dec_w1 = (const float*)d_in[16];
  const float* dec_b1 = (const float*)d_in[17];
  const float* dec_w2 = (const float*)d_in[18];
  const float* dec_b2 = (const float*)d_in[19];
  const float* dec_w3 = (const float*)d_in[20];
  const float* dec_b3 = (const float*)d_in[21];
  const float* li_w1  = (const float*)d_in[22];
  const float* li_b1  = (const float*)d_in[23];
  const float* li_w2  = (const float*)d_in[24];
  const float* li_b2  = (const float*)d_in[25];

  // workspace layout (floats)
  float* ws = (float*)d_ws;
  long long off = 0;
  float* q       = ws + off; off += (long long)MM * DA;        // 8192x128
  float* kbuf    = ws + off; off += (long long)MM * DA;        // 8192x128
  float* v       = ws + off; off += (long long)MM * DD;        // 8192x256
  float* S       = ws + off; off += (long long)BB * NN * NN;   // 16x512x512
  float* attn    = ws + off; off += (long long)MM * DD;
  float* z       = ws + off; off += (long long)MM * DD;
  float* ffn_out = ws + off; off += (long long)MM * DD;
  float* z_coor  = ws + off; off += (long long)MM * DD;
  float* z_local = ws + off; off += (long long)MM * DD;
  float* quant   = ws + off; off += (long long)MM * DD;
  float* xrec    = ws + off; off += (long long)MM * DD;
  float* H1      = ws + off; off += (long long)MM * HH;        // 8192x1024
  float* H2      = ws + off; off += (long long)MM * HH;

  float* psi    = (float*)d_out;
  float* d_loss = psi + PSI_ELEMS;
  (void)hipMemsetAsync(d_loss, 0, sizeof(float), stream);

  const dim3 blk(128);
  const int nElem = MM * DD;                 // 2,097,152
  const float invN = 1.f / (float)nElem;

  // --- projections (M=8192) ---
  gemm_bf16_wmma<0, false><<<dim3(DA / 64, MM / 64, 1), blk, 0, stream>>>(
      x, DD, 0, Wq_w, DA, 0, Wq_b, q, DA, 0, DD);
  gemm_bf16_wmma<0, false><<<dim3(DA / 64, MM / 64, 1), blk, 0, stream>>>(
      x, DD, 0, Wk_w, DA, 0, Wk_b, kbuf, DA, 0, DD);
  gemm_bf16_wmma<0, false><<<dim3(DD / 64, MM / 64, 1), blk, 0, stream>>>(
      x, DD, 0, Wv_w, DD, 0, Wv_b, v, DD, 0, DD);

  // --- attention scores: S_b = q_b @ k_b^T (batched over 16) ---
  gemm_bf16_wmma<0, true><<<dim3(NN / 64, NN / 64, BB), blk, 0, stream>>>(
      q, DA, (long long)NN * DA,
      kbuf, DA, (long long)NN * DA,
      nullptr, S, NN, (long long)NN * NN, DA);

  softmax_rows<<<MM, 256, 0, stream>>>(S, 0.08838834764831845f /* 1/sqrt(128) */);

  // --- attn_out = P @ v (batched) ---
  gemm_bf16_wmma<0, false><<<dim3(DD / 64, NN / 64, BB), blk, 0, stream>>>(
      S, NN, (long long)NN * NN,
      v, DD, (long long)NN * DD,
      nullptr, attn, DD, (long long)NN * DD, NN);

  ln_residual<<<MM, 256, 0, stream>>>(x, attn, ln1_g, ln1_b, z);

  // --- FFN ---
  gemm_bf16_wmma<1, false><<<dim3(HH / 64, MM / 64, 1), blk, 0, stream>>>(
      z, DD, 0, ffn_w1, HH, 0, ffn_b1, H1, HH, 0, DD);
  gemm_bf16_wmma<0, false><<<dim3(DD / 64, MM / 64, 1), blk, 0, stream>>>(
      H1, HH, 0, ffn_w2, DD, 0, ffn_b2, ffn_out, DD, 0, HH);

  ln_residual<<<MM, 256, 0, stream>>>(z, ffn_out, ln2_g, ln2_b, z_coor);

  // --- local inference head ---
  gemm_bf16_wmma<1, false><<<dim3(HH / 64, MM / 64, 1), blk, 0, stream>>>(
      x, DD, 0, li_w1, HH, 0, li_b1, H1, HH, 0, DD);
  gemm_bf16_wmma<0, false><<<dim3(DD / 64, MM / 64, 1), blk, 0, stream>>>(
      H1, HH, 0, li_w2, DD, 0, li_b2, z_local, DD, 0, HH);
  mse_acc<<<1024, 256, 0, stream>>>(z_coor, z_local, nElem, invN, d_loss);

  // --- VQ: argmin + gather; vq_loss = 1.25 * mse(quant, lat) ---
  vq_kernel<<<MM, 64, 0, stream>>>(z_coor, emb, quant);
  mse_acc<<<1024, 256, 0, stream>>>(quant, z_coor, nElem, 1.25f * invN, d_loss);

  // --- decoder (z_quant forward value == quant) ---
  gemm_bf16_wmma<1, false><<<dim3(HH / 64, MM / 64, 1), blk, 0, stream>>>(
      quant, DD, 0, dec_w1, HH, 0, dec_b1, H1, HH, 0, DD);
  gemm_bf16_wmma<1, false><<<dim3(HH / 64, MM / 64, 1), blk, 0, stream>>>(
      H1, HH, 0, dec_w2, HH, 0, dec_b2, H2, HH, 0, HH);
  gemm_bf16_wmma<0, false><<<dim3(DD / 64, MM / 64, 1), blk, 0, stream>>>(
      H2, HH, 0, dec_w3, DD, 0, dec_b3, xrec, DD, 0, HH);
  mse_acc<<<1024, 256, 0, stream>>>(xrec, x, nElem, invN, d_loss);

  // --- psi_hid broadcast/concat (1.07 GB streaming write) ---
  psi_kernel<<<MM, 128, 0, stream>>>(z_coor, emb, psi);
}
